// GraphDiscriminator_89550068122064
// MI455X (gfx1250) — matrix operand order repaired and checked by reading, not compiled, not run on hardware
//
#include <hip/hip_runtime.h>

// ---------------- problem constants (match reference) ----------------
#define NN      10000
#define EE      160000
#define NGG     64
#define HIDC    256
#define MAXH    4
#define NEG_SLOPE 0.2f

// ---------------- WMMA vector types (gfx1250, wave32) ----------------
typedef __attribute__((ext_vector_type(16))) __bf16 v16bf;
typedef __attribute__((ext_vector_type(8)))  __bf16 v8bf;
typedef __attribute__((ext_vector_type(8)))  float  v8f;
typedef __attribute__((ext_vector_type(4)))  unsigned int u32x4;
typedef __attribute__((ext_vector_type(8)))  int  i32x8;
typedef __attribute__((ext_vector_type(4)))  int  i32x4;

#if __has_builtin(__builtin_amdgcn_tensor_load_to_lds)
#define HAVE_TDM 1
#else
#define HAVE_TDM 0
#endif

// round-to-nearest-even f32 -> bf16 (bit pattern)
__device__ __forceinline__ unsigned short f32_to_bf16(float f) {
    unsigned u = __float_as_uint(f);
    unsigned r = u + 0x7FFFu + ((u >> 16) & 1u);
    return (unsigned short)(r >> 16);
}

__device__ __forceinline__ float atomicMaxF(float* addr, float v) {
    if (v >= 0.0f)
        return __int_as_float(atomicMax((int*)addr, __float_as_int(v)));
    else
        return __uint_as_float(atomicMin((unsigned int*)addr, __float_as_uint(v)));
}

__device__ __forceinline__ void wait_tensor0() {
#if HAVE_TDM
 #if __has_builtin(__builtin_amdgcn_s_wait_tensorcnt)
    __builtin_amdgcn_s_wait_tensorcnt(0);
 #else
    asm volatile("s_wait_tensorcnt 0x0" ::: "memory");
 #endif
#endif
}

#if HAVE_TDM
// Issue a TDM DMA: 64 rows x 32 halfs (bf16) tile of BT[Nout][K] -> compacted LDS tile.
// D# per CDNA5 ISA ch.8: group0 = count/lds_addr/global_addr/type, group1 = sizes/strides.
__device__ __forceinline__ void tdm_stage_b(const unsigned short* gsrc, unsigned lds_off, int K) {
    unsigned long long ga = (unsigned long long)(const char*)gsrc;
    u32x4 g0;
    g0.x = 1u;                                                     // count=1, user mode
    g0.y = (unsigned)__builtin_amdgcn_readfirstlane((int)lds_off); // lds_addr (bytes)
    g0.z = (unsigned)__builtin_amdgcn_readfirstlane((int)(unsigned)(ga & 0xFFFFFFFFull));
    g0.w = (unsigned)__builtin_amdgcn_readfirstlane(
               (int)(((unsigned)(ga >> 32) & 0x01FFFFFFu) | (2u << 30))); // addr hi | type=2
    const unsigned TD = 1u << 20;      // oversized tensor dims: tile always in-bounds
    i32x8 g1;
    g1[0] = 0x10000;                                   // data_size=1 -> 2-byte elements
    g1[1] = (int)((TD & 0xFFFFu) << 16);               // tensor_dim0[15:0] (atomic bar addr=0)
    g1[2] = (int)(((TD >> 16) & 0xFFFFu) | ((TD & 0xFFFFu) << 16)); // td0 hi | td1 lo
    g1[3] = (int)(((TD >> 16) & 0xFFFFu) | (32u << 16));            // td1 hi | tile_dim0=32
    g1[4] = 64;                                        // tile_dim1=64 rows, tile_dim2=0
    g1[5] = __builtin_amdgcn_readfirstlane(K);         // tensor_dim0_stride[31:0] = K
    g1[6] = 0;                                         // stride hi | dim1_stride lo
    g1[7] = 0;
    i32x4 gz = {0, 0, 0, 0};
#if __clang_major__ >= 23
    i32x8 gz8 = {0, 0, 0, 0, 0, 0, 0, 0};
    __builtin_amdgcn_tensor_load_to_lds(g0, g1, gz, gz, gz8, 0);
#else
    __builtin_amdgcn_tensor_load_to_lds(g0, g1, gz, gz, 0);
#endif
}
#endif

// ---------------- generic fill ----------------
__global__ void k_fill_f32(float* __restrict__ p, long long n, float v) {
    long long i = (long long)blockIdx.x * blockDim.x + threadIdx.x;
    if (i < n) p[i] = v;
}

// ---------------- cast X (f32 [n,K]) -> bf16 [n,Kpad], zero-pad K ----------------
__global__ void k_cast_x_bf16(const float* __restrict__ X, unsigned short* __restrict__ xb,
                              int n, int K, int Kpad) {
    long long i = (long long)blockIdx.x * blockDim.x + threadIdx.x;
    if (i >= (long long)n * Kpad) return;
    int kp  = (int)(i % Kpad);
    long long row = i / Kpad;
    xb[i] = (kp < K) ? f32_to_bf16(X[row * K + kp]) : (unsigned short)0;
}

// ---------------- W (f32 [K,Nout]) -> WT bf16 [Nout,Kpad], zero-pad K ----------------
__global__ void k_w_transpose_bf16(const float* __restrict__ W, unsigned short* __restrict__ wt,
                                   int K, int Kpad, int Nout) {
    long long i = (long long)blockIdx.x * blockDim.x + threadIdx.x;
    if (i >= (long long)Nout * Kpad) return;
    int kp   = (int)(i % Kpad);
    int nout = (int)(i / Kpad);
    wt[i] = (kp < K) ? f32_to_bf16(W[(long long)kp * Nout + nout]) : (unsigned short)0;
}

// ---------------- bf16 WMMA GEMM: C[M,Nout] = A[M,K] * BT[Nout,K]^T ----------------
// Block = 8 waves covering a 128(M) x 64(N) tile. The 64-col B panel K-slab
// (64 rows x 32 halfs = 4KB) is staged into LDS once per block per K-step by
// the Tensor Data Mover (double-buffered, s_wait_tensorcnt), shared by all 8
// waves; each wave runs 4x v_wmma_f32_16x16x32_bf16 per slab.
__global__ void __launch_bounds__(256)
k_gemm_bf16_wmma(const unsigned short* __restrict__ A,
                 const unsigned short* __restrict__ BT,
                 float* __restrict__ C,
                 int M, int Nout, int K) {
    __shared__ __attribute__((aligned(16))) unsigned short sB[2][64 * 32];

    int tid    = threadIdx.x;
    int waveId = tid >> 5;
    int lane   = tid & 31;
    int mtiles = M >> 4;
    int mb     = (mtiles + 7) >> 3;             // M-tile groups of 8 per block
    int bn     = blockIdx.x / mb;
    int bm     = blockIdx.x - bn * mb;
    int mt     = (bm << 3) + waveId;
    bool active = (mt < mtiles);                // wave-uniform: EXEC stays all-1s
    int m0 = mt << 4;
    int n0 = bn << 6;

    // A 16x32 as loaded here (contiguous k, same order as B): lanes 0-15 -> M=lane, halves=K+0..15
    //                        lanes 16-31 -> same M, K+16..31
    int row   = m0 + (lane & 15);
    int koffA = (lane < 16) ? 0 : 16;
    // B 32x16 as loaded here (contiguous k, same order as A): lanes 0-15 -> N=lane, halves=K+0..15; lanes 16-31 -> K+16..31
    int koffB = (lane < 16) ? 0 : 16;
    int nlane = lane & 15;

    const __bf16* Ab = (const __bf16*)A;
    v8f acc[4] = {v8f{}, v8f{}, v8f{}, v8f{}};

    // ---- prologue: stage K-slab 0 into buffer 0 ----
#if HAVE_TDM
    if (waveId == 0)
        tdm_stage_b(BT + (size_t)n0 * K, (unsigned)(size_t)(void*)&sB[0][0], K);
    wait_tensor0();
#else
    {   // cooperative fallback: 256 threads x 16B
        int r = tid >> 2, c = (tid & 3) << 3;
        *(v8bf*)&sB[0][r * 32 + c] = *(const v8bf*)(BT + (size_t)(n0 + r) * K + c);
    }
#endif
    __syncthreads();

    int cur = 0;
    for (int k0 = 0; k0 < K; k0 += 32) {
        int nxt = cur ^ 1;
#if HAVE_TDM
        if (k0 + 32 < K && waveId == 0)
            tdm_stage_b(BT + (size_t)n0 * K + (k0 + 32),
                        (unsigned)(size_t)(void*)&sB[nxt][0], K);
#endif
        if (active) {
            const __bf16* ap = Ab + (size_t)row * K + k0 + koffA;
            v8bf alo = *(const v8bf*)(ap);
            v8bf ahi = *(const v8bf*)(ap + 8);
            v16bf afrag = __builtin_shufflevector(alo, ahi,
                            0,1,2,3,4,5,6,7,8,9,10,11,12,13,14,15);
            __builtin_prefetch(ap + 32, 0, 0);      // global_prefetch_b8: next K-slab of A

#pragma unroll
            for (int t = 0; t < 4; ++t) {
                const __bf16* bp = (const __bf16*)&sB[cur][(t * 16 + nlane) * 32 + koffB];
                v8bf blo = *(const v8bf*)(bp);      // ds_load_b128
                v8bf bhi = *(const v8bf*)(bp + 8);  // ds_load_b128
                v16bf bfrag = __builtin_shufflevector(blo, bhi,
                                0,1,2,3,4,5,6,7,8,9,10,11,12,13,14,15);
                acc[t] = __builtin_amdgcn_wmma_f32_16x16x32_bf16(
                            false, afrag, false, bfrag, (short)0, acc[t], false, false);
            }
        }
#if !HAVE_TDM
        if (k0 + 32 < K) {
            int r = tid >> 2, c = (tid & 3) << 3;
            *(v8bf*)&sB[nxt][r * 32 + c] =
                *(const v8bf*)(BT + (size_t)(n0 + r) * K + (k0 + 32) + c);
        }
#endif
        wait_tensor0();
        __syncthreads();
        cur = nxt;
    }

    // 32-bit C/D 16x16: VGPR r -> M=r (lanes 0-15) / 8+r (lanes 16-31); N=lane&15
    if (active) {
        int rofs = (lane < 16) ? 0 : 8;
#pragma unroll
        for (int t = 0; t < 4; ++t) {
            int col = n0 + t * 16 + nlane;
#pragma unroll
            for (int r = 0; r < 8; ++r) {
                C[(size_t)(m0 + r + rofs) * Nout + col] = acc[t][r];
            }
        }
    }
}

// ---------------- attention scores: a_s[n,h] = sum_c h[n,h,c]*att_s[h,c] ----------------
__global__ void k_attn_scores(const float* __restrict__ H,
                              const float* __restrict__ attS, const float* __restrict__ attD,
                              float* __restrict__ a_s, float* __restrict__ a_d,
                              int n, int heads) {
    int gw   = (int)((blockIdx.x * blockDim.x + threadIdx.x) >> 5);
    int lane = threadIdx.x & 31;
    if (gw >= n * heads) return;
    int node = gw / heads;
    int h    = gw % heads;
    const float* hp = H + (size_t)node * heads * HIDC + (size_t)h * HIDC;
    float ss = 0.0f, sd = 0.0f;
    for (int j = lane; j < HIDC; j += 32) {
        float v = hp[j];
        ss += v * attS[h * HIDC + j];
        sd += v * attD[h * HIDC + j];
    }
    for (int o = 16; o > 0; o >>= 1) {
        ss += __shfl_down(ss, o, 32);
        sd += __shfl_down(sd, o, 32);
    }
    if (lane == 0) { a_s[gw] = ss; a_d[gw] = sd; }
}

// ---------------- edge logits (leaky relu) + segment max ----------------
__global__ void k_edge_logits(const int* __restrict__ ei,
                              const float* __restrict__ a_s, const float* __restrict__ a_d,
                              float* __restrict__ elog, float* __restrict__ m,
                              int E_, int Eext, int heads) {
    long long idx = (long long)blockIdx.x * blockDim.x + threadIdx.x;
    if (idx >= (long long)Eext * heads) return;
    int e = (int)(idx / heads);
    int h = (int)(idx % heads);
    int s = (e < E_) ? ei[e]       : (e - E_);
    int d = (e < E_) ? ei[E_ + e]  : (e - E_);
    float l = a_s[s * heads + h] + a_d[d * heads + h];
    l = (l > 0.0f) ? l : NEG_SLOPE * l;
    elog[idx] = l;
    atomicMaxF(&m[d * heads + h], l);
}

// ---------------- edge exp + segment sum ----------------
__global__ void k_edge_exp(const int* __restrict__ ei,
                           float* __restrict__ elog, const float* __restrict__ m,
                           float* __restrict__ den, int E_, int Eext, int heads) {
    long long idx = (long long)blockIdx.x * blockDim.x + threadIdx.x;
    if (idx >= (long long)Eext * heads) return;
    int e = (int)(idx / heads);
    int h = (int)(idx % heads);
    int d = (e < E_) ? ei[E_ + e] : (e - E_);
    float mv = m[d * heads + h];
    if (mv <= -3.0e38f) mv = 0.0f;                  // isfinite guard (empty segments)
    float ex = expf(elog[idx] - mv);
    elog[idx] = ex;
    atomicAdd(&den[d * heads + h], ex);
}

// ---------------- scatter: out[dst] += alpha * h[src] ----------------
__global__ void k_scatter(const int* __restrict__ ei,
                          const float* __restrict__ elog, const float* __restrict__ den,
                          const float* __restrict__ H, float* __restrict__ OUT,
                          int E_, int heads, int Nout) {
    int e = blockIdx.x;
    int s = (e < E_) ? ei[e]      : (e - E_);
    int d = (e < E_) ? ei[E_ + e] : (e - E_);
    const float* hs = H   + (size_t)s * Nout;
    float*       od = OUT + (size_t)d * Nout;
    for (int i = threadIdx.x; i < Nout; i += blockDim.x) {
        int hh = i >> 8;   // HIDC == 256
        float alpha = elog[(size_t)e * heads + hh] / (den[d * heads + hh] + 1e-16f);
        atomicAdd(&od[i], alpha * hs[i]);
    }
}

// ---------------- bias + ELU ----------------
__global__ void k_bias_elu(const float* __restrict__ OUT, const float* __restrict__ bias,
                           float* __restrict__ X, long long total, int Nout) {
    long long i = (long long)blockIdx.x * blockDim.x + threadIdx.x;
    if (i >= total) return;
    float v = OUT[i] + bias[(int)(i % Nout)];
    X[i] = (v > 0.0f) ? v : (expf(v) - 1.0f);
}

// ---------------- global mean pool (atomic accumulate) ----------------
__global__ void k_pool(const float* __restrict__ X, const int* __restrict__ batch,
                       float* __restrict__ sums, float* __restrict__ counts) {
    int n = blockIdx.x;
    int g = batch[n];
    for (int i = threadIdx.x; i < HIDC; i += blockDim.x)
        atomicAdd(&sums[(size_t)g * HIDC + i], X[(size_t)n * HIDC + i]);
    if (threadIdx.x == 0) atomicAdd(&counts[g], 1.0f);
}

// ---------------- features = sums / max(counts,1); also write to d_out ----------------
__global__ void k_features(float* __restrict__ sums, const float* __restrict__ counts,
                           float* __restrict__ out_feat) {
    int i = blockIdx.x * blockDim.x + threadIdx.x;
    if (i >= NGG * HIDC) return;
    int g = i / HIDC;
    float f = sums[i] / fmaxf(counts[g], 1.0f);
    sums[i] = f;            // in-place: sums becomes `features` for fc1
    out_feat[i] = f;
}

// ---------------- MLP head ----------------
__global__ void k_fc1(const float* __restrict__ feat, const float* __restrict__ w,
                      const float* __restrict__ b, float* __restrict__ hmid) {
    int g = blockIdx.x, t = threadIdx.x;          // 64 blocks x 128 threads
    float acc = b[t];
    const float* fr = feat + (size_t)g * HIDC;
    for (int c = 0; c < HIDC; ++c) acc += fr[c] * w[c * (HIDC / 2) + t];
    hmid[g * (HIDC / 2) + t] = fmaxf(acc, 0.0f);
}

__global__ void k_fc2(const float* __restrict__ hmid, const float* __restrict__ w,
                      const float* __restrict__ b, float* __restrict__ validity) {
    int g = threadIdx.x;                           // 1 block x 64 threads
    if (g >= NGG) return;
    float acc = b[0];
    const float* hr = hmid + (size_t)g * (HIDC / 2);
    for (int c = 0; c < HIDC / 2; ++c) acc += hr[c] * w[c];
    validity[g] = acc;
}

// =====================================================================
extern "C" void kernel_launch(void* const* d_in, const int* in_sizes, int n_in,
                              void* d_out, int out_size, void* d_ws, size_t ws_size,
                              hipStream_t stream) {
    const float* atom  = (const float*)d_in[0];
    const int*   ei    = (const int*)  d_in[1];
    const int*   batch = (const int*)  d_in[2];
    const float* W1 = (const float*)d_in[3];
    const float* as1 = (const float*)d_in[4];
    const float* ad1 = (const float*)d_in[5];
    const float* b1  = (const float*)d_in[6];
    const float* W2 = (const float*)d_in[7];
    const float* as2 = (const float*)d_in[8];
    const float* ad2 = (const float*)d_in[9];
    const float* b2  = (const float*)d_in[10];
    const float* W3 = (const float*)d_in[11];
    const float* as3 = (const float*)d_in[12];
    const float* ad3 = (const float*)d_in[13];
    const float* b3  = (const float*)d_in[14];
    const float* fc1w = (const float*)d_in[15];
    const float* fc1b = (const float*)d_in[16];
    const float* fc2w = (const float*)d_in[17];
    const float* fc2b = (const float*)d_in[18];
    float* out = (float*)d_out;

    const int Eext = EE + NN;
    const long long BIG = (long long)NN * 1024;

    // ---- carve workspace (256B aligned blocks) ----
    char* w = (char*)d_ws;
    size_t off = 0;
    auto alloc = [&](size_t bytes) -> char* {
        char* p = w + off;
        off = (off + bytes + 255) & ~(size_t)255;
        return p;
    };
    float* Xbuf   = (float*)alloc(BIG * 4);                 // activations between layers
    float* Hbuf   = (float*)alloc(BIG * 4);                 // GEMM output h
    float* OUTbuf = (float*)alloc(BIG * 4);                 // scatter accumulator
    unsigned short* xb = (unsigned short*)alloc(BIG * 2);   // bf16 activations
    unsigned short* wt = (unsigned short*)alloc((size_t)1024 * 1024 * 2); // bf16 W^T
    float* a_s  = (float*)alloc((size_t)NN * MAXH * 4);
    float* a_d  = (float*)alloc((size_t)NN * MAXH * 4);
    float* mbuf = (float*)alloc((size_t)NN * MAXH * 4);
    float* den  = (float*)alloc((size_t)NN * MAXH * 4);
    float* elog = (float*)alloc((size_t)Eext * MAXH * 4);
    float* sums   = (float*)alloc((size_t)NGG * HIDC * 4);
    float* counts = (float*)alloc((size_t)NGG * 4);
    float* hmid   = (float*)alloc((size_t)NGG * (HIDC / 2) * 4);

    auto nblk = [](long long total, int bs) { return (unsigned)((total + bs - 1) / bs); };

    auto run_layer = [&](const float* Xin, int K, int Kpad, int Nout, int heads,
                         const float* Wm, const float* attS, const float* attD,
                         const float* bias, float* Xout) {
        long long nk = (long long)NN * Kpad;
        k_cast_x_bf16<<<nblk(nk, 256), 256, 0, stream>>>(Xin, xb, NN, K, Kpad);
        long long wk = (long long)Nout * Kpad;
        k_w_transpose_bf16<<<nblk(wk, 256), 256, 0, stream>>>(Wm, wt, K, Kpad, Nout);

        int mtiles = NN / 16;                       // 625
        int mb     = (mtiles + 7) / 8;              // 79 blocks per N-group
        int ng     = Nout / 64;
        k_gemm_bf16_wmma<<<mb * ng, 256, 0, stream>>>(xb, wt, Hbuf, NN, Nout, Kpad);

        int aw = NN * heads;
        k_attn_scores<<<nblk(aw, 8), 256, 0, stream>>>(Hbuf, attS, attD, a_s, a_d, NN, heads);

        k_fill_f32<<<nblk(aw, 256), 256, 0, stream>>>(mbuf, aw, -3.402823466e38f);
        k_fill_f32<<<nblk(aw, 256), 256, 0, stream>>>(den, aw, 0.0f);

        long long ew = (long long)Eext * heads;
        k_edge_logits<<<nblk(ew, 256), 256, 0, stream>>>(ei, a_s, a_d, elog, mbuf, EE, Eext, heads);
        k_edge_exp<<<nblk(ew, 256), 256, 0, stream>>>(ei, elog, mbuf, den, EE, Eext, heads);

        long long no = (long long)NN * Nout;
        k_fill_f32<<<nblk(no, 256), 256, 0, stream>>>(OUTbuf, no, 0.0f);
        k_scatter<<<Eext, 256, 0, stream>>>(ei, elog, den, Hbuf, OUTbuf, EE, heads, Nout);
        k_bias_elu<<<nblk(no, 256), 256, 0, stream>>>(OUTbuf, bias, Xout, no, Nout);
    };

    // GAT layer 1: 23 -> 1024 (4 heads x 256)
    run_layer(atom, 23, 32, MAXH * HIDC, MAXH, W1, as1, ad1, b1, Xbuf);
    // GAT layer 2: 1024 -> 1024
    run_layer(Xbuf, MAXH * HIDC, MAXH * HIDC, MAXH * HIDC, MAXH, W2, as2, ad2, b2, Xbuf);
    // GAT layer 3: 1024 -> 256 (1 head)
    run_layer(Xbuf, MAXH * HIDC, MAXH * HIDC, HIDC, 1, W3, as3, ad3, b3, Xbuf);

    // global mean pool -> features
    k_fill_f32<<<nblk(NGG * HIDC, 256), 256, 0, stream>>>(sums, NGG * HIDC, 0.0f);
    k_fill_f32<<<1, 256, 0, stream>>>(counts, NGG, 0.0f);
    k_pool<<<NN, 256, 0, stream>>>(Xbuf, batch, sums, counts);
    k_features<<<nblk(NGG * HIDC, 256), 256, 0, stream>>>(sums, counts, out + NGG);

    // MLP head -> validity
    k_fc1<<<NGG, HIDC / 2, 0, stream>>>(sums, fc1w, fc1b, hmid);
    k_fc2<<<1, 64, 0, stream>>>(hmid, fc2w, fc2b, out);
}
